// NormEMAVectorQuantizer_5729486373542
// MI455X (gfx1250) — matrix operand order repaired and checked by reading, not compiled, not run on hardware
//
#include <hip/hip_runtime.h>
#include <hip/hip_bf16.h>

typedef __attribute__((ext_vector_type(16))) _Float16 v16h;
typedef __attribute__((ext_vector_type(8)))  _Float16 v8h;
typedef __attribute__((ext_vector_type(8)))  float    v8f;

constexpr int   D_  = 32;      // channels
constexpr int   K_  = 8192;    // codes
constexpr int   N_  = 16384;   // B*H*W = 16*32*32
constexpr int   HW_ = 1024;    // 32*32
constexpr float DECAY_ = 0.99f;

// ---------------------------------------------------------------------------
// 1) channel-last + l2norm; emit f32 and f16 copies of zn
// ---------------------------------------------------------------------------
__global__ void vq_prep_zn(const float* __restrict__ z,
                           float* __restrict__ zn32,
                           _Float16* __restrict__ zn16) {
  int n = blockIdx.x * blockDim.x + threadIdx.x;
  if (n >= N_) return;
  int b = n / HW_, hw = n % HW_;
  const float* src = z + (size_t)b * D_ * HW_ + hw;
  float v[D_]; float ss = 0.f;
#pragma unroll
  for (int d = 0; d < D_; ++d) { v[d] = src[(size_t)d * HW_]; ss += v[d] * v[d]; }
  float inv = 1.f / fmaxf(sqrtf(ss), 1e-12f);
#pragma unroll
  for (int d = 0; d < D_; ++d) {
    float x = v[d] * inv;
    zn32[(size_t)n * D_ + d] = x;
    zn16[(size_t)n * D_ + d] = (_Float16)x;
  }
}

// ---------------------------------------------------------------------------
// 2) codebook -> f16, zero counts/sums/loss accumulator
// ---------------------------------------------------------------------------
__global__ void vq_prep_emb(const float* __restrict__ emb,
                            _Float16* __restrict__ emb16,
                            float* __restrict__ counts,
                            float* __restrict__ sums,
                            float* __restrict__ loss_acc) {
  int i = blockIdx.x * blockDim.x + threadIdx.x;
  if (i < K_ * D_) { emb16[i] = (_Float16)emb[i]; sums[i] = 0.f; }
  if (i < K_)      counts[i] = 0.f;
  if (i == 0)      loss_acc[0] = 0.f;
}

// ---------------------------------------------------------------------------
// 3) WMMA cosine-sim + argmax. One block = one 16-row M tile, 8 waves split K.
//    v_wmma_f32_16x16x32_f16: A = zn rows (16x32), B = 16 codes (32x16).
//    K loop unrolled x2 so the 2nd WMMA + loads fill the D->VALU hazard window.
// ---------------------------------------------------------------------------
__global__ void __launch_bounds__(256)
vq_argmax(const _Float16* __restrict__ zn16,
          const _Float16* __restrict__ emb16,
          int* __restrict__ ids,
          float* __restrict__ out_ids) {
  __shared__ float svals[8][16];
  __shared__ int   sidx [8][16];

  const int lane = threadIdx.x & 31;
  const int wave = threadIdx.x >> 5;
  const int hi   = lane >> 4;      // half-wave select
  const int lp   = lane & 15;
  const int r0   = blockIdx.x * 16;

  // A fragment (contiguous K order, matches B):
  // lane<16: row=r0+lp, K dims {0..15}; lane>=16: dims {16..31}
  union { v16h v; v8h h[2]; } a;
  const _Float16* arow = zn16 + (size_t)(r0 + lp) * D_ + 16 * hi;
  a.h[0] = *(const v8h*)(arow);
  a.h[1] = *(const v8h*)(arow + 8);

  float bestv[8]; int besti[8];
#pragma unroll
  for (int j = 0; j < 8; ++j) { bestv[j] = -3.0e38f; besti[j] = 0x7fffffff; }

  const int tilesPerWave = (K_ / 16) / 8;  // 64
  const int t0 = wave * tilesPerWave;
  // B fragment base for this lane: code = c0 + lp, dims 16*hi..16*hi+15
  const _Float16* bbase = emb16 + (size_t)(t0 * 16 + lp) * D_ + 16 * hi;

  for (int t = 0; t < tilesPerWave; t += 2) {
    const _Float16* bp0 = bbase + (size_t)t * 16 * D_;
    const _Float16* bp1 = bp0 + 16 * D_;
    __builtin_prefetch(bp1 + 16 * D_, 0, 3);       // global_prefetch (t+2)
    __builtin_prefetch(bp1 + 32 * D_, 0, 3);       // global_prefetch (t+3)
    v16h bm0 = *(const v16h*)bp0;
    v16h bm1 = *(const v16h*)bp1;
    v8f cz0 = {};
    v8f cz1 = {};
    v8f c0 = __builtin_amdgcn_wmma_f32_16x16x32_f16(
        false, a.v, false, bm0, (short)0, cz0, false, false);
    v8f c1 = __builtin_amdgcn_wmma_f32_16x16x32_f16(
        false, a.v, false, bm1, (short)0, cz1, false, false);
    const int code0 = (t0 + t) * 16 + lp;          // N = lane&15
    const int code1 = code0 + 16;
    // strict > keeps lowest index on ties (codes ascend within the wave)
#pragma unroll
    for (int j = 0; j < 8; ++j) {                  // c[j]: row r0 + j + 8*hi
      float v0 = c0[j];
      if (v0 > bestv[j]) { bestv[j] = v0; besti[j] = code0; }
    }
#pragma unroll
    for (int j = 0; j < 8; ++j) {
      float v1 = c1[j];
      if (v1 > bestv[j]) { bestv[j] = v1; besti[j] = code1; }
    }
  }

  // reduce over the 16 codes held across lanes of each half-wave
#pragma unroll
  for (int j = 0; j < 8; ++j) {
    float v = bestv[j]; int i = besti[j];
#pragma unroll
    for (int m = 8; m >= 1; m >>= 1) {   // xor 8,4,2,1 stays within half-wave
      float ov = __shfl_xor(v, m, 32);
      int   oi = __shfl_xor(i, m, 32);
      if (ov > v || (ov == v && oi < i)) { v = ov; i = oi; }
    }
    bestv[j] = v; besti[j] = i;
  }
  if (lp == 0) {   // lane 0 -> rows j, lane 16 -> rows j+8
#pragma unroll
    for (int j = 0; j < 8; ++j) {
      svals[wave][hi * 8 + j] = bestv[j];
      sidx [wave][hi * 8 + j] = besti[j];
    }
  }
  __syncthreads();

  if (threadIdx.x < 16) {
    int row = threadIdx.x;
    float v = svals[0][row]; int i = sidx[0][row];
#pragma unroll
    for (int w = 1; w < 8; ++w) {
      float ov = svals[w][row]; int oi = sidx[w][row];
      if (ov > v || (ov == v && oi < i)) { v = ov; i = oi; }
    }
    ids[r0 + row]     = i;
    out_ids[r0 + row] = (float)i;   // d_out is float-typed
  }
}

// ---------------------------------------------------------------------------
// 4) scatter: counts + per-dim sums (lane-per-dim -> consecutive-addr atomics)
// ---------------------------------------------------------------------------
__global__ void vq_scatter(const int* __restrict__ ids,
                           const float* __restrict__ zn32,
                           float* __restrict__ counts,
                           float* __restrict__ sums) {
  int i = blockIdx.x * blockDim.x + threadIdx.x;   // over N*D
  if (i >= N_ * D_) return;
  int n = i >> 5, d = i & 31;
  int id = ids[n];
  atomicAdd(&sums[(size_t)id * D_ + d], zn32[i]);
  if (d == 0) atomicAdd(&counts[id], 1.0f);
}

// ---------------------------------------------------------------------------
// 5) EMA update of codebook + usage
// ---------------------------------------------------------------------------
__global__ void vq_update(const float* __restrict__ emb,
                          const float* __restrict__ counts,
                          const float* __restrict__ sums,
                          const float* __restrict__ code_usage,
                          float* __restrict__ out_emb,
                          float* __restrict__ out_usage) {
  int k = blockIdx.x * blockDim.x + threadIdx.x;
  if (k >= K_) return;
  float cnt = counts[k];
  out_usage[k] = DECAY_ * code_usage[k] + (1.0f - DECAY_) * cnt;

  float e[D_], m[D_];
  float denom = fmaxf(cnt, 1.0f);
  float ss = 0.f;
#pragma unroll
  for (int d = 0; d < D_; ++d) {
    e[d] = emb[(size_t)k * D_ + d];
    float sel = (cnt > 0.f) ? sums[(size_t)k * D_ + d] / denom : e[d];
    m[d] = sel; ss += sel * sel;
  }
  float inv = 1.f / fmaxf(sqrtf(ss), 1e-12f);
  float ss2 = 0.f;
#pragma unroll
  for (int d = 0; d < D_; ++d) {
    float t = DECAY_ * e[d] + (1.0f - DECAY_) * (m[d] * inv);
    m[d] = t; ss2 += t * t;
  }
  float inv2 = 1.f / fmaxf(sqrtf(ss2), 1e-12f);
#pragma unroll
  for (int d = 0; d < D_; ++d) out_emb[(size_t)k * D_ + d] = m[d] * inv2;
}

// ---------------------------------------------------------------------------
// 6) z_q gather back to channel-first (straight-through value == embedding[id])
// ---------------------------------------------------------------------------
__global__ void vq_zq(const int* __restrict__ ids,
                      const float* __restrict__ emb,
                      float* __restrict__ out) {
  int i = blockIdx.x * blockDim.x + threadIdx.x;   // over B*D*H*W
  if (i >= N_ * D_) return;
  int b  = i / (D_ * HW_);
  int r  = i % (D_ * HW_);
  int d  = r / HW_;
  int hw = r % HW_;
  int n  = b * HW_ + hw;
  out[i] = emb[(size_t)ids[n] * D_ + d];
}

// ---------------------------------------------------------------------------
// 7) cosine embedding loss (target=1): wave-reduce then one atomic per wave
// ---------------------------------------------------------------------------
__global__ void vq_loss(const int* __restrict__ ids,
                        const float* __restrict__ emb,
                        const float* __restrict__ zn32,
                        float* __restrict__ acc) {
  int n = blockIdx.x * blockDim.x + threadIdx.x;
  float l = 0.f;
  if (n < N_) {
    int id = ids[n];
    float dot = 0.f, na = 0.f, nb = 0.f;
#pragma unroll
    for (int d = 0; d < D_; ++d) {
      float av = emb[(size_t)id * D_ + d];
      float bv = zn32[(size_t)n * D_ + d];
      dot += av * bv; na += av * av; nb += bv * bv;
    }
    l = 1.f - dot / fmaxf(sqrtf(na) * sqrtf(nb), 1e-8f);
  }
#pragma unroll
  for (int m = 16; m >= 1; m >>= 1) l += __shfl_xor(l, m, 32);
  if ((threadIdx.x & 31) == 0) atomicAdd(acc, l);
}

__global__ void vq_finish(const float* __restrict__ acc, float* __restrict__ out_loss) {
  out_loss[0] = acc[0] / (float)N_;
}

// ---------------------------------------------------------------------------
extern "C" void kernel_launch(void* const* d_in, const int* in_sizes, int n_in,
                              void* d_out, int out_size, void* d_ws, size_t ws_size,
                              hipStream_t stream) {
  const float* z          = (const float*)d_in[0];   // [16,32,32,32]
  const float* embedding  = (const float*)d_in[1];   // [8192,32]
  const float* code_usage = (const float*)d_in[2];   // [8192]
  (void)in_sizes; (void)n_in; (void)out_size; (void)ws_size;

  // ---- output layout (floats) ----
  float* out      = (float*)d_out;
  float* o_zq     = out;                               // 524288
  float* o_ids    = out + (size_t)N_ * D_;             // 16384
  float* o_loss   = o_ids + N_;                        // 1
  float* o_emb    = o_loss + 1;                        // 262144
  float* o_usage  = o_emb + (size_t)K_ * D_;           // 8192

  // ---- workspace layout ----
  char* ws = (char*)d_ws;
  float*    zn32   = (float*)(ws);                              // 2 MB
  _Float16* zn16   = (_Float16*)(ws + 2097152);                 // 1 MB
  _Float16* emb16  = (_Float16*)(ws + 2097152 + 1048576);       // 512 KB
  int*      ids    = (int*)   (ws + 3670016);                   // 64 KB
  float*    counts = (float*) (ws + 3735552);                   // 32 KB
  float*    sums   = (float*) (ws + 3768320);                   // 1 MB
  float*    lacc   = (float*) (ws + 4816896);                   // 4 B

  vq_prep_zn  <<<(N_ + 255) / 256, 256, 0, stream>>>(z, zn32, zn16);
  vq_prep_emb <<<(K_ * D_ + 255) / 256, 256, 0, stream>>>(embedding, emb16, counts, sums, lacc);
  vq_argmax   <<<N_ / 16, 256, 0, stream>>>(zn16, emb16, ids, o_ids);
  vq_scatter  <<<(N_ * D_ + 255) / 256, 256, 0, stream>>>(ids, zn32, counts, sums);
  vq_update   <<<(K_ + 255) / 256, 256, 0, stream>>>(embedding, counts, sums, code_usage, o_emb, o_usage);
  vq_zq       <<<(N_ * D_ + 255) / 256, 256, 0, stream>>>(ids, embedding, o_zq);
  vq_loss     <<<(N_ + 255) / 256, 256, 0, stream>>>(ids, embedding, zn32, lacc);
  vq_finish   <<<1, 1, 0, stream>>>(lacc, o_loss);
}
